// MHA_90271622628098
// MI455X (gfx1250) — matrix operand order repaired and checked by reading, not compiled, not run on hardware
//
#include <hip/hip_runtime.h>

// ============================================================================
// MHA forward (B=2, L=2048, D=1024, H=16, DK=64) for MI455X / gfx1250.
// Outputs (concatenated in d_out): y [B,L,D] f32, A [B,H,L,L] f32.
//
// ~69 GFLOP vs 537MB A-matrix traffic (written raw once, read+rewritten once)
// -> balanced at dense bf16 WMMA rates, so every GEMM stage uses
// v_wmma_f32_16x16x32_bf16 with f32 accumulation. All WMMA operand loads are
// contiguous 128-bit accesses: weights are pre-transposed to bf16 Wt[n][k]
// (Q scale folded in), V is stored transposed Vt[dk][l]. The A*V kernel
// double-buffers V tiles into LDS with global_load_async_to_lds_b128 /
// s_wait_asynccnt so the HBM V stream overlaps the exp+WMMA work.
//
// Workspace (~48.5 MB):
//   Qb  bf16 [B,H,L,DK]      Kb bf16 [B,H,L,DK]     Vt bf16 [B,H,DK,L]
//   Wt  bf16 [4][D][D]       (transposed weights; Wt_q pre-scaled by 1/8)
//   rowmax/rowinv f32 [B*H*L]
//   O   f32 [B,L,D]
// ============================================================================

constexpr int Bn  = 2;
constexpr int Ln  = 2048;
constexpr int Dn  = 1024;
constexpr int Hn  = 16;
constexpr int DKn = 64;
constexpr int NQT = Ln / 16;   // 128 tiles along L
constexpr int BH  = Bn * Hn;   // 32

typedef __bf16 v16bf __attribute__((ext_vector_type(16)));
typedef float  v8f   __attribute__((ext_vector_type(8)));

static __device__ __forceinline__ v8f wmma_bf16(v16bf a, v16bf b, v8f c) {
  return __builtin_amdgcn_wmma_f32_16x16x32_bf16(false, a, false, b, (short)0, c,
                                                 false, false);
}

// ---- fragment loaders (contiguous k order, same in A and B, wave32) ----------------
// A-operand 16x32 (MxK) 16-bit: lane l holds row m=l&15;
//   e[i] = K = k0+(l>>4)*16+i
static __device__ __forceinline__ v16bf load_a_f32(const float* row, int lane, int k0) {
  const int kb = (lane >> 4) * 16;
  const float* p0 = row + k0 + kb;
  const float* p1 = row + k0 + 8 + kb;
  v16bf a;
#pragma unroll
  for (int i = 0; i < 8; ++i) a[i] = (__bf16)p0[i];
#pragma unroll
  for (int i = 0; i < 8; ++i) a[8 + i] = (__bf16)p1[i];
  return a;
}

static __device__ __forceinline__ v16bf load_a_bf16(const __bf16* row, int lane, int k0) {
  const int kb = (lane >> 4) * 16;
  const __bf16* p0 = row + k0 + kb;
  const __bf16* p1 = row + k0 + 8 + kb;
  v16bf a;
#pragma unroll
  for (int i = 0; i < 8; ++i) a[i] = p0[i];
#pragma unroll
  for (int i = 0; i < 8; ++i) a[8 + i] = p1[i];
  return a;
}

// B-operand 32x16 (KxN) 16-bit: lane l holds column n=l&15; e[i]=K=(l>>4)*16+i
// -> 16 contiguous bf16 when the source is K-major.
static __device__ __forceinline__ v16bf load_b_contig(const __bf16* p) {
  v16bf b;
#pragma unroll
  for (int i = 0; i < 16; ++i) b[i] = p[i];
  return b;
}

// ============================================================================
// K0: weight prep. f32 W[k][n] -> bf16 Wt[n][k] (LDS transpose, padded).
// z: 0=Wq (scaled by 1/sqrt(DK)) 1=Wk 2=Wv 3=Wo. grid=(64,64,4), block=256.
// ============================================================================
__global__ __launch_bounds__(256) void wprep_kernel(
    const float* __restrict__ Wq, const float* __restrict__ Wk,
    const float* __restrict__ Wv, const float* __restrict__ Wo,
    __bf16* __restrict__ Wt) {
  __shared__ float tile[16][17];
  const int z = blockIdx.z;
  const float* W = (z == 0) ? Wq : (z == 1) ? Wk : (z == 2) ? Wv : Wo;
  __bf16* T = Wt + (size_t)z * Dn * Dn;
  const int tx = threadIdx.x & 15, ty = threadIdx.x >> 4;
  const int k0 = blockIdx.x * 16, n0 = blockIdx.y * 16;
  tile[ty][tx] = W[(size_t)(k0 + ty) * Dn + (n0 + tx)];
  __syncthreads();
  const float s = (z == 0) ? 0.125f : 1.0f;
  T[(size_t)(n0 + ty) * Dn + (k0 + tx)] = (__bf16)(tile[tx][ty] * s);
}

// ============================================================================
// K1: fused QKV projection. grid=(256 row-tiles, 64 col-tiles, 3 weights)
// ============================================================================
__global__ __launch_bounds__(32) void qkv_proj_kernel(
    const float* __restrict__ X, const __bf16* __restrict__ Wt,
    __bf16* __restrict__ Qb, __bf16* __restrict__ Kb, __bf16* __restrict__ Vt) {
  const int lane = threadIdx.x;
  const int rt = blockIdx.x, ct = blockIdx.y, z = blockIdx.z;
  const int half = lane >> 4;

  const float*  xrow = X + ((size_t)rt * 16 + (lane & 15)) * Dn;
  const __bf16* wrow = Wt + (size_t)z * Dn * Dn + (size_t)(ct * 16 + (lane & 15)) * Dn;
  v8f c = {};
  for (int k0 = 0; k0 < Dn; k0 += 32) {
    v16bf a = load_a_f32(xrow, lane, k0);
    v16bf b = load_b_contig(wrow + k0 + half * 16);   // contiguous 32B
    c = wmma_bf16(a, b, c);
  }

  const int j  = ct * 16 + (lane & 15);
  const int h  = j >> 6, dk = j & 63;
#pragma unroll
  for (int r = 0; r < 8; ++r) {
    const int row = rt * 16 + 8 * half + r;
    const int bb = row >> 11;
    const int lq = row & (Ln - 1);
    const __bf16 val = (__bf16)c[r];
    const size_t bh = (size_t)bb * Hn + h;
    if (z == 2)       Vt[(bh * DKn + dk) * Ln + lq] = val;
    else if (z == 0)  Qb[(bh * Ln + lq) * DKn + dk] = val;
    else              Kb[(bh * Ln + lq) * DKn + dk] = val;
  }
}

// ============================================================================
// K2: raw masked scores S = Q K^T. grid=(128 ktiles, 128 qtiles, 32 bh)
// ============================================================================
__global__ __launch_bounds__(32) void scores_kernel(
    const __bf16* __restrict__ Qb, const __bf16* __restrict__ Kb,
    float* __restrict__ Araw) {
  const int lane = threadIdx.x;
  const int kt = blockIdx.x, qt = blockIdx.y, bh = blockIdx.z;
  const int n = lane & 15, half = lane >> 4;
  float* Ab = Araw + (size_t)bh * Ln * Ln;

  if (kt > qt) {
#pragma unroll
    for (int r = 0; r < 8; ++r) {
      const int q = qt * 16 + 8 * half + r;
      Ab[(size_t)q * Ln + kt * 16 + n] = -1e9f;
    }
    return;
  }

  const __bf16* qrow = Qb + ((size_t)bh * Ln + qt * 16 + n) * DKn;
  const __bf16* krow = Kb + ((size_t)bh * Ln + kt * 16 + n) * DKn;
  v8f c = {};
#pragma unroll
  for (int s = 0; s < 2; ++s) {
    v16bf a = load_a_bf16(qrow, lane, s * 32);
    v16bf b = load_b_contig(krow + s * 32 + half * 16);
    c = wmma_bf16(a, b, c);
  }
#pragma unroll
  for (int r = 0; r < 8; ++r) {
    const int q = qt * 16 + 8 * half + r;
    const int k = kt * 16 + n;
    Ab[(size_t)q * Ln + k] = (k > q) ? -1e9f : c[r];
  }
}

// ============================================================================
// K3: per-row max and 1/sum(exp). one wave per row, 8 waves per block.
// ============================================================================
__global__ __launch_bounds__(256) void rowstats_kernel(
    const float* __restrict__ Araw, float* __restrict__ rowmax,
    float* __restrict__ rowinv) {
  const int lane = threadIdx.x & 31;
  const size_t row = (size_t)blockIdx.x * 8 + (threadIdx.x >> 5);
  const float* p = Araw + row * Ln;

  float m = -3.0e38f;
  for (int j = lane; j < Ln; j += 32) m = fmaxf(m, p[j]);
#pragma unroll
  for (int off = 16; off; off >>= 1) m = fmaxf(m, __shfl_xor(m, off, 32));

  float s = 0.0f;
  for (int j = lane; j < Ln; j += 32) s += __expf(p[j] - m);
#pragma unroll
  for (int off = 16; off; off >>= 1) s += __shfl_xor(s, off, 32);

  if (lane == 0) { rowmax[row] = m; rowinv[row] = 1.0f / s; }
}

// ============================================================================
// K4: fused softmax-normalize (writes final A) + O = A*V, with V tiles
// double-buffered into LDS via async global->LDS DMA (ASYNCcnt pipeline).
// grid = (128 qtiles, 32 bh), 1 wave per workgroup.
// ============================================================================
__global__ __launch_bounds__(32) void av_kernel(
    float* __restrict__ A, const __bf16* __restrict__ Vt,
    const float* __restrict__ rowmax, const float* __restrict__ rowinv,
    float* __restrict__ O) {
  // vtile[buf][dk][kk] : 32-key x 64-dk bf16 tile (4KB), double buffered.
  __shared__ __align__(16) __bf16 vtile[2][DKn][32];

  const int lane = threadIdx.x;
  const int qt = blockIdx.x, bh = blockIdx.y;
  const int n = lane & 15, half = lane >> 4;
  const int b = bh >> 4, h = bh & 15;

  float* Ab = A + (size_t)bh * Ln * Ln;
  const int qa = qt * 16 + n;                 // this lane's A-operand row
  float* arow = Ab + (size_t)qa * Ln;
  const float m   = rowmax[(size_t)bh * Ln + qa];
  const float inv = rowinv[(size_t)bh * Ln + qa];
  const int ksend = (qt * 16 + 16 + 31) & ~31;   // first fully-masked slice

  const char* vbytes = (const char*)(Vt + (size_t)bh * DKn * Ln);

  // Stage one 4KB V tile (keys [ks,ks+32) x dk 0..63) into vtile[buf]:
  // 8 x global_load_async_to_lds_b128, 16 bytes per lane per op.
  auto stage = [&](int buf, int ks) {
    const unsigned lbase = (unsigned)(size_t)&vtile[buf][0][0];
#pragma unroll
    for (int j = 0; j < 8; ++j) {
      const int slot = j * 32 + lane;           // 0..255 16B-slots
      const int dk   = slot >> 2;               // 4 slots per 64B key-row
      const int bo   = (slot & 3) * 16;
      const unsigned long long gaddr =
          (unsigned long long)(size_t)(vbytes + ((size_t)dk * Ln + ks) * 2 + bo);
      const unsigned laddr = lbase + (unsigned)(dk * 64 + bo);
      asm volatile("global_load_async_to_lds_b128 %0, %1, off"
                   :: "v"(laddr), "v"(gaddr) : "memory");
    }
  };

  stage(0, 0);

  v8f c0 = {}, c1 = {}, c2 = {}, c3 = {};
  for (int ks = 0; ks < ksend; ks += 32) {
    const int buf = (ks >> 5) & 1;
    if (ks + 32 < ksend) {
      stage(buf ^ 1, ks + 32);                  // prefetch next tile
      asm volatile("s_wait_asynccnt 0x8" ::: "memory");  // current tile done
    } else {
      asm volatile("s_wait_asynccnt 0x0" ::: "memory");
    }

    float* p0 = arow + ks + half * 16;
    float* p1 = arow + ks + 8 + half * 16;
    __builtin_prefetch(arow + ks + 32, 0, 0);   // global_prefetch_b8 (A stream)

    v16bf a;
#pragma unroll
    for (int i = 0; i < 8; ++i) {
      const float pv = __expf(p0[i] - m) * inv;
      p0[i] = pv;                               // final A element
      a[i] = (__bf16)pv;
    }
#pragma unroll
    for (int i = 0; i < 8; ++i) {
      const float pv = __expf(p1[i] - m) * inv;
      p1[i] = pv;
      a[8 + i] = (__bf16)pv;
    }

    // B frags from LDS: e[i] = V[ks + half*16 + i][nb*16 + n]
    c0 = wmma_bf16(a, load_b_contig(&vtile[buf][0 * 16 + n][half * 16]), c0);
    c1 = wmma_bf16(a, load_b_contig(&vtile[buf][1 * 16 + n][half * 16]), c1);
    c2 = wmma_bf16(a, load_b_contig(&vtile[buf][2 * 16 + n][half * 16]), c2);
    c3 = wmma_bf16(a, load_b_contig(&vtile[buf][3 * 16 + n][half * 16]), c3);
  }

  // causal tail: final A is exactly zero there
  for (int ks = ksend; ks < Ln; ks += 32) {
    float* p0 = arow + ks + half * 8;
    float* p1 = arow + ks + 16 + half * 8;
#pragma unroll
    for (int i = 0; i < 8; ++i) { p0[i] = 0.0f; p1[i] = 0.0f; }
  }

#pragma unroll
  for (int r = 0; r < 8; ++r) {
    const int q = qt * 16 + 8 * half + r;
    const size_t orow = ((size_t)b * Ln + q) * Dn + h * DKn;
    O[orow + 0 * 16 + n] = c0[r];
    O[orow + 1 * 16 + n] = c1[r];
    O[orow + 2 * 16 + n] = c2[r];
    O[orow + 3 * 16 + n] = c3[r];
  }
}

// ============================================================================
// K5: final projection y = O * W_o^T-layout. grid=(256,64)
// ============================================================================
__global__ __launch_bounds__(32) void oproj_kernel(
    const float* __restrict__ O, const __bf16* __restrict__ Wt,
    float* __restrict__ Y) {
  const int lane = threadIdx.x;
  const int rt = blockIdx.x, ct = blockIdx.y;
  const int half = lane >> 4;
  const float*  orow = O + ((size_t)rt * 16 + (lane & 15)) * Dn;
  const __bf16* wrow = Wt + (size_t)3 * Dn * Dn + (size_t)(ct * 16 + (lane & 15)) * Dn;
  v8f c = {};
  for (int k0 = 0; k0 < Dn; k0 += 32) {
    v16bf a = load_a_f32(orow, lane, k0);
    v16bf b = load_b_contig(wrow + k0 + half * 16);
    c = wmma_bf16(a, b, c);
  }
#pragma unroll
  for (int r = 0; r < 8; ++r) {
    const size_t row = (size_t)rt * 16 + 8 * half + r;
    Y[row * Dn + ct * 16 + (lane & 15)] = c[r];
  }
}

// ============================================================================
extern "C" void kernel_launch(void* const* d_in, const int* in_sizes, int n_in,
                              void* d_out, int out_size, void* d_ws, size_t ws_size,
                              hipStream_t stream) {
  (void)in_sizes; (void)n_in; (void)out_size; (void)ws_size;
  const float* X  = (const float*)d_in[0];
  // d_in[1] = causal mask (triu, k=1) -- known statically, not read.
  const float* Wq = (const float*)d_in[2];
  const float* Wk = (const float*)d_in[3];
  const float* Wv = (const float*)d_in[4];
  const float* Wo = (const float*)d_in[5];

  float* Y = (float*)d_out;                                  // [B,L,D]
  float* A = (float*)d_out + (size_t)Bn * Ln * Dn;           // [B,H,L,L]

  constexpr size_t QKV_ELEMS = (size_t)Bn * Hn * Ln * DKn;   // 4,194,304
  constexpr size_t ROWS      = (size_t)Bn * Hn * Ln;         // 65,536
  char* ws = (char*)d_ws;
  __bf16* Qb     = (__bf16*)ws;  ws += QKV_ELEMS * sizeof(__bf16);
  __bf16* Kb     = (__bf16*)ws;  ws += QKV_ELEMS * sizeof(__bf16);
  __bf16* Vt     = (__bf16*)ws;  ws += QKV_ELEMS * sizeof(__bf16);
  __bf16* Wt     = (__bf16*)ws;  ws += (size_t)4 * Dn * Dn * sizeof(__bf16);
  float*  rowmax = (float*)ws;   ws += ROWS * sizeof(float);
  float*  rowinv = (float*)ws;   ws += ROWS * sizeof(float);
  float*  O      = (float*)ws;   ws += (size_t)Bn * Ln * Dn * sizeof(float);

  // K0: transpose+convert weights (Q scale folded in)
  wprep_kernel<<<dim3(Dn / 16, Dn / 16, 4), dim3(256), 0, stream>>>(
      Wq, Wk, Wv, Wo, Wt);

  // K1: QKV projections
  qkv_proj_kernel<<<dim3(Bn * Ln / 16, Dn / 16, 3), dim3(32), 0, stream>>>(
      X, Wt, Qb, Kb, Vt);

  // K2: raw masked scores into A region
  scores_kernel<<<dim3(NQT, NQT, BH), dim3(32), 0, stream>>>(Qb, Kb, A);

  // K3: per-row softmax statistics
  rowstats_kernel<<<dim3((unsigned)(ROWS / 8)), dim3(256), 0, stream>>>(
      A, rowmax, rowinv);

  // K4: normalize A in place + O = A*V (async-LDS pipelined V)
  av_kernel<<<dim3(NQT, BH), dim3(32), 0, stream>>>(A, Vt, rowmax, rowinv, O);

  // K5: y = O * W_o
  oproj_kernel<<<dim3(Bn * Ln / 16, Dn / 16), dim3(32), 0, stream>>>(O, Wt, Y);
}
